// DaiJazz_Piano_37572373905992
// MI455X (gfx1250) — matrix operand order repaired and checked by reading, not compiled, not run on hardware
//
#include <hip/hip_runtime.h>
#include <hip/hip_bf16.h>

typedef __hip_bfloat16 bf_t;
typedef __attribute__((ext_vector_type(16))) __bf16 bf16x16;
typedef __attribute__((ext_vector_type(8)))  __bf16 bf16x8;
typedef __attribute__((ext_vector_type(8)))  float  f32x8;
typedef int v4i __attribute__((vector_size(16)));   // matches builtin param type

union BFrag { bf16x16 v; bf16x8 h[2]; };

constexpr int kB = 4, kT = 512, kM = kB * kT;          // 2048 token rows
constexpr int kE = 2130, kEP = 2176;                   // embed dim, padded (17*128)
constexpr int kH = 10, kD = 213, kDP = 256;            // heads, head dim, padded (2*128)
constexpr int kBH = kB * kH;                           // 40
constexpr int kFF = 4 * kE, kFFP = 8576;               // 8520 -> 67*128
constexpr int kQKVN = 3 * kE, kQKVP = 6528;            // 6390 -> 51*128
constexpr int kV = 512, kRN = 2 * kT - 1, kL = 6;      // vocab, rel rows, layers

// ---------------------------------------------------------------- WMMA helper
__device__ __forceinline__ f32x8 wmma_bf16f32(bf16x16 a, bf16x16 b, f32x8 c) {
#if defined(__gfx1250__)
  return __builtin_amdgcn_wmma_f32_16x16x32_bf16(false, a, false, b, (short)0, c,
                                                 false, false);
#else
  return c;   // host pass never executes this
#endif
}

// ------------------------------------------------- async global->LDS helpers
#if defined(__gfx1250__) && __has_builtin(__builtin_amdgcn_global_load_async_to_lds_b128)
#define USE_ASYNC_LDS 1
#else
#define USE_ASYNC_LDS 0
#endif

// copy 16 bytes (8 bf16) global -> LDS, per lane
__device__ __forceinline__ void async_copy16(const bf_t* g, bf_t* l) {
#if USE_ASYNC_LDS
  __builtin_amdgcn_global_load_async_to_lds_b128(
      (__attribute__((address_space(1))) v4i*)(unsigned long long)(const void*)g,
      (__attribute__((address_space(3))) v4i*)(unsigned)(unsigned long long)(void*)l,
      0, 0);
#else
  *(uint4*)l = *(const uint4*)g;
#endif
}

__device__ __forceinline__ void wait_async0() {
#if USE_ASYNC_LDS
#if __has_builtin(__builtin_amdgcn_s_wait_asynccnt)
  __builtin_amdgcn_s_wait_asynccnt(0);
#else
  asm volatile("s_wait_asynccnt 0x0" ::: "memory");
#endif
#endif
}

// ---------------------------------------------------------------- reductions
__device__ __forceinline__ float warpSum(float v) {
  for (int o = 16; o > 0; o >>= 1) v += __shfl_down(v, o, 32);
  return v;
}
__device__ __forceinline__ float warpMax(float v) {
  for (int o = 16; o > 0; o >>= 1) v = fmaxf(v, __shfl_down(v, o, 32));
  return v;
}
__device__ float blockSum(float v) {
  __shared__ float sh[32];
  int lane = threadIdx.x & 31, wid = threadIdx.x >> 5;
  v = warpSum(v);
  if (lane == 0) sh[wid] = v;
  __syncthreads();
  int nw = (blockDim.x + 31) >> 5;
  v = (threadIdx.x < nw) ? sh[threadIdx.x] : 0.f;
  if (wid == 0) v = warpSum(v);
  if (threadIdx.x == 0) sh[0] = v;
  __syncthreads();
  v = sh[0];
  __syncthreads();
  return v;
}
__device__ float blockMax(float v) {
  __shared__ float sh[32];
  int lane = threadIdx.x & 31, wid = threadIdx.x >> 5;
  v = warpMax(v);
  if (lane == 0) sh[wid] = v;
  __syncthreads();
  int nw = (blockDim.x + 31) >> 5;
  v = (threadIdx.x < nw) ? sh[threadIdx.x] : -3.4e38f;
  if (wid == 0) v = warpMax(v);
  if (threadIdx.x == 0) sh[0] = v;
  __syncthreads();
  v = sh[0];
  __syncthreads();
  return v;
}

// =================================================================== GEMM
// C(f32, MxN) = A(bf16, MxK, row major, lda) * B(bf16, stored N x K, ldb)
// M,N multiples of 128; K multiple of 32. 256 thr = 8 waves; wave = 32x64.
// Double-buffered LDS; staging via async global->LDS (ASYNCcnt).
__global__ __launch_bounds__(256) void gemm_bf16_128x128(
    const bf_t* __restrict__ A, const bf_t* __restrict__ Bm,
    float* __restrict__ C, int Kdim, int lda, int ldb, int ldc,
    long long strideA, long long strideB, long long strideC) {
  const int batch = blockIdx.z;
  A  += (size_t)batch * strideA;
  Bm += (size_t)batch * strideB;
  C  += (size_t)batch * strideC;

  __shared__ __align__(16) bf_t As[2][128 * 32];
  __shared__ __align__(16) bf_t Bs[2][128 * 32];

  const int tid = threadIdx.x, wid = tid >> 5, lane = tid & 31;
  const int wm = wid >> 1, wn = wid & 1;
  const int bm = blockIdx.y * 128, bn = blockIdx.x * 128;

  const f32x8 zero = {0.f, 0.f, 0.f, 0.f, 0.f, 0.f, 0.f, 0.f};
  f32x8 acc[2][4];
  for (int i = 0; i < 2; ++i)
    for (int j = 0; j < 4; ++j) acc[i][j] = zero;

  const int lrow = tid >> 1, lcol = (tid & 1) * 16;   // 16 bf16 per thread
  const bf_t* Agp = A + (size_t)(bm + lrow) * lda + lcol;
  const bf_t* Bgp = Bm + (size_t)(bn + lrow) * ldb + lcol;

  // prologue: stage K-step 0 into buffer 0
  {
    bf_t* al = &As[0][lrow * 32 + lcol];
    bf_t* bl = &Bs[0][lrow * 32 + lcol];
    async_copy16(Agp, al);
    async_copy16(Agp + 8, al + 8);
    async_copy16(Bgp, bl);
    async_copy16(Bgp + 8, bl + 8);
  }
  wait_async0();
  __syncthreads();

  const int nsteps = Kdim >> 5;
  const int arow = wm * 32 + (lane & 15);
  const int akb  = (lane < 16) ? 0 : 8;               // A frag K base (ISA layout)
  const int bcol = wn * 64 + (lane & 15);
  const int bkb  = (lane < 16) ? 0 : 8;               // B frag K base

  for (int s = 0; s < nsteps; ++s) {
    const int cur = s & 1, nxt = cur ^ 1;
    if (s + 1 < nsteps) {                              // prefetch next K tile
      const bf_t* ag = Agp + (size_t)(s + 1) * 32;
      const bf_t* bg = Bgp + (size_t)(s + 1) * 32;
      bf_t* al = &As[nxt][lrow * 32 + lcol];
      bf_t* bl = &Bs[nxt][lrow * 32 + lcol];
      async_copy16(ag, al);
      async_copy16(ag + 8, al + 8);
      async_copy16(bg, bl);
      async_copy16(bg + 8, bl + 8);
    }
    if (s + 2 < nsteps) {                              // L2 prefetch 2 ahead
      __builtin_prefetch(Agp + (size_t)(s + 2) * 32, 0, 0);
      __builtin_prefetch(Bgp + (size_t)(s + 2) * 32, 0, 0);
    }

    const bf_t* Ac = As[cur];
    const bf_t* Bc = Bs[cur];
    BFrag a[2], b[4];
    a[0].h[0] = *(const bf16x8*)&Ac[arow * 32 + akb];
    a[0].h[1] = *(const bf16x8*)&Ac[arow * 32 + akb + 16];
    a[1].h[0] = *(const bf16x8*)&Ac[(arow + 16) * 32 + akb];
    a[1].h[1] = *(const bf16x8*)&Ac[(arow + 16) * 32 + akb + 16];
#pragma unroll
    for (int j = 0; j < 4; ++j) {
      b[j].h[0] = *(const bf16x8*)&Bc[(bcol + j * 16) * 32 + bkb];
      b[j].h[1] = *(const bf16x8*)&Bc[(bcol + j * 16) * 32 + bkb + 16];
    }
#pragma unroll
    for (int i = 0; i < 2; ++i)
#pragma unroll
      for (int j = 0; j < 4; ++j)
        acc[i][j] = wmma_bf16f32(a[i].v, b[j].v, acc[i][j]);

    wait_async0();      // next buffer resident before anyone proceeds
    __syncthreads();
  }

  const int rh = (lane >> 4) * 8, cc = lane & 15;     // C layout per ISA
#pragma unroll
  for (int i = 0; i < 2; ++i)
#pragma unroll
    for (int j = 0; j < 4; ++j)
#pragma unroll
      for (int r = 0; r < 8; ++r) {
        int row = bm + wm * 32 + i * 16 + rh + r;
        int col = bn + wn * 64 + j * 16 + cc;
        C[(size_t)row * ldc + col] = acc[i][j][r];
      }
}

// ======================================= BD: S[bh,x,y] += qr[bh,x,:] . R[x,y,:]
// M = 40 (pad 48), N-block = 64, K = 256.  grid(8, 512), 128 threads.
__global__ __launch_bounds__(128) void bd_wmma_kernel(
    const bf_t* __restrict__ qr,   // (BH, T, DP)
    const bf_t* __restrict__ rbf,  // (RN, DP)
    float* __restrict__ S) {       // (BH, T, T)
  const int x = blockIdx.y, bn = blockIdx.x * 64;
  __shared__ __align__(16) bf_t As[48 * 32];
  __shared__ __align__(16) bf_t Bs[64 * 32];
  const int tid = threadIdx.x, wid = tid >> 5, lane = tid & 31;

  // zero the A pad rows (40..47) once; async copies never touch them
  {
    uint4 zz = {0, 0, 0, 0};
    for (int z = tid; z < 32; z += 128) ((uint4*)&As[40 * 32])[z] = zz;
  }

  const f32x8 zero = {0.f, 0.f, 0.f, 0.f, 0.f, 0.f, 0.f, 0.f};
  f32x8 acc[3] = {zero, zero, zero};

  const int akb = (lane < 16) ? 0 : 8;
  const int bcol = wid * 16 + (lane & 15);
  const int bkb = (lane < 16) ? 0 : 8;

  for (int k0 = 0; k0 < kDP; k0 += 32) {
    if (tid < 80) {                       // A: 40 valid rows x 32
      int r = tid >> 1, c = (tid & 1) * 16;
      const bf_t* p = qr + ((size_t)r * kT + x) * kDP + k0 + c;
      bf_t* dst = &As[r * 32 + c];
      async_copy16(p, dst);
      async_copy16(p + 8, dst + 8);
    }
    {                                     // B: rows y, data = r_emb[T-1+x-y]
      int r = tid >> 1, c = (tid & 1) * 16;
      int y = bn + r;
      const bf_t* p = rbf + (size_t)(kT - 1 + x - y) * kDP + k0 + c;
      bf_t* dst = &Bs[r * 32 + c];
      async_copy16(p, dst);
      async_copy16(p + 8, dst + 8);
    }
    wait_async0();
    __syncthreads();
    BFrag b;
    b.h[0] = *(const bf16x8*)&Bs[bcol * 32 + bkb];
    b.h[1] = *(const bf16x8*)&Bs[bcol * 32 + bkb + 16];
#pragma unroll
    for (int i = 0; i < 3; ++i) {
      BFrag a;
      int ar = i * 16 + (lane & 15);
      a.h[0] = *(const bf16x8*)&As[ar * 32 + akb];
      a.h[1] = *(const bf16x8*)&As[ar * 32 + akb + 16];
      acc[i] = wmma_bf16f32(a.v, b.v, acc[i]);
    }
    __syncthreads();
  }
  const int rh = (lane >> 4) * 8, cc = lane & 15;
#pragma unroll
  for (int i = 0; i < 3; ++i)
#pragma unroll
    for (int r = 0; r < 8; ++r) {
      int m = i * 16 + rh + r;            // bh index
      if (m < kBH) {
        size_t off = ((size_t)m * kT + x) * kT + bn + wid * 16 + cc;
        S[off] += acc[i][r];
      }
    }
}

// ---------------------------------------------------------------- elementwise
__global__ void embed_kernel(const int* __restrict__ idx,
                             const float* __restrict__ tok,
                             const float* __restrict__ pos,
                             float* __restrict__ xf) {
  size_t i = (size_t)blockIdx.x * blockDim.x + threadIdx.x;
  if (i >= (size_t)kM * kEP) return;
  int c = (int)(i % kEP);
  int row = (int)(i / kEP);
  float v = 0.f;
  if (c < kE) v = tok[(size_t)idx[row] * kE + c] + pos[(size_t)(row & (kT - 1)) * kE + c];
  xf[i] = v;
}

__global__ void ln_bf16_kernel(const float* __restrict__ x,
                               const float* __restrict__ g,
                               const float* __restrict__ b,
                               bf_t* __restrict__ out) {
  int row = blockIdx.x;
  const float* xr = x + (size_t)row * kEP;
  float s = 0.f;
  for (int c = threadIdx.x; c < kE; c += blockDim.x) s += xr[c];
  float mu = blockSum(s) * (1.f / kE);
  float vs = 0.f;
  for (int c = threadIdx.x; c < kE; c += blockDim.x) {
    float d = xr[c] - mu;
    vs += d * d;
  }
  float rstd = rsqrtf(blockSum(vs) * (1.f / kE) + 1e-5f);
  for (int c = threadIdx.x; c < kEP; c += blockDim.x) {
    float v = (c < kE) ? (xr[c] - mu) * rstd * g[c] + b[c] : 0.f;
    out[(size_t)row * kEP + c] = __float2bfloat16(v);
  }
}

// f32 (K x N) -> bf16 (Np x Kp) transposed + zero pad
__global__ void convT_kernel(const float* __restrict__ src, bf_t* __restrict__ dst,
                             int K, int N, int Kp, int Np) {
  size_t i = (size_t)blockIdx.x * blockDim.x + threadIdx.x;
  if (i >= (size_t)Np * Kp) return;
  int k = (int)(i % Kp);
  int n = (int)(i / Kp);
  float v = (n < N && k < K) ? src[(size_t)k * N + n] : 0.f;
  dst[i] = __float2bfloat16(v);
}

__global__ void remb_kernel(const float* __restrict__ r, bf_t* __restrict__ dst) {
  size_t i = (size_t)blockIdx.x * blockDim.x + threadIdx.x;
  if (i >= (size_t)kRN * kDP) return;
  int d = (int)(i & (kDP - 1));
  int row = (int)(i >> 8);
  dst[i] = __float2bfloat16(d < kD ? r[(size_t)row * kD + d] : 0.f);
}

__global__ void split_qkv_kernel(const float* __restrict__ qkv,
                                 const float* __restrict__ rw,
                                 const float* __restrict__ rr,
                                 bf_t* __restrict__ qw, bf_t* __restrict__ qr,
                                 bf_t* __restrict__ kb, bf_t* __restrict__ vT) {
  size_t i = (size_t)blockIdx.x * blockDim.x + threadIdx.x;
  if (i >= (size_t)kBH * kT * kDP) return;
  int d = (int)(i & (kDP - 1));
  size_t j = i >> 8;
  int t = (int)(j & (kT - 1));
  int bh = (int)(j >> 9);
  int b = bh / kH, h = bh % kH;
  float q = 0.f, k = 0.f, v = 0.f, w = 0.f, r = 0.f;
  if (d < kD) {
    size_t ro = (size_t)(b * kT + t) * kQKVP;
    q = qkv[ro + h * kD + d];
    k = qkv[ro + kE + h * kD + d];
    v = qkv[ro + 2 * kE + h * kD + d];
    w = rw[h * kD + d];
    r = rr[h * kD + d];
  }
  size_t o = ((size_t)bh * kT + t) * kDP + d;
  bool ok = d < kD;
  qw[o] = __float2bfloat16(ok ? q + w : 0.f);
  qr[o] = __float2bfloat16(ok ? q + r : 0.f);
  kb[o] = __float2bfloat16(ok ? k : 0.f);
  vT[((size_t)bh * kDP + d) * kT + t] = __float2bfloat16(ok ? v : 0.f);
}

__global__ void softmax_kernel(const float* __restrict__ S, bf_t* __restrict__ P) {
  size_t row = blockIdx.x;                       // (bh, x) flattened
  const float* s = S + row * kT;
  float m = -3.4e38f;
  for (int c = threadIdx.x; c < kT; c += blockDim.x) m = fmaxf(m, s[c]);
  m = blockMax(m);
  float sum = 0.f;
  for (int c = threadIdx.x; c < kT; c += blockDim.x) sum += __expf(s[c] - m);
  sum = blockSum(sum);
  float inv = 1.f / sum;
  for (int c = threadIdx.x; c < kT; c += blockDim.x)
    P[row * kT + c] = __float2bfloat16(__expf(s[c] - m) * inv);
}

__global__ void merge_o_kernel(const float* __restrict__ of, bf_t* __restrict__ obf) {
  size_t i = (size_t)blockIdx.x * blockDim.x + threadIdx.x;
  if (i >= (size_t)kM * kEP) return;
  int c = (int)(i % kEP);
  int row = (int)(i / kEP);
  float v = 0.f;
  if (c < kE) {
    int h = c / kD, d = c % kD;
    int b = row >> 9, t = row & (kT - 1);
    v = of[((size_t)(b * kH + h) * kT + t) * kDP + d];
  }
  obf[i] = __float2bfloat16(v);
}

__global__ void add_kernel(float* __restrict__ x, const float* __restrict__ y, size_t n) {
  size_t i = (size_t)blockIdx.x * blockDim.x + threadIdx.x;
  if (i < n) x[i] += y[i];
}

__global__ void bias_relu_bf16_kernel(const float* __restrict__ f,
                                      const float* __restrict__ b1,
                                      bf_t* __restrict__ o) {
  size_t i = (size_t)blockIdx.x * blockDim.x + threadIdx.x;
  if (i >= (size_t)kM * kFFP) return;
  int c = (int)(i % kFFP);
  float v = (c < kFF) ? fmaxf(f[i] + b1[c], 0.f) : 0.f;
  o[i] = __float2bfloat16(v);
}

__global__ void add_bias_res_kernel(float* __restrict__ x, const float* __restrict__ y,
                                    const float* __restrict__ b2) {
  size_t i = (size_t)blockIdx.x * blockDim.x + threadIdx.x;
  if (i >= (size_t)kM * kEP) return;
  int c = (int)(i % kEP);
  x[i] += y[i] + (c < kE ? b2[c] : 0.f);
}

__global__ void head_bias_kernel(float* __restrict__ o, const float* __restrict__ b) {
  size_t i = (size_t)blockIdx.x * blockDim.x + threadIdx.x;
  if (i >= (size_t)kM * kV) return;
  o[i] += b[i % kV];
}

__global__ void nll_rows_kernel(const float* __restrict__ logits,
                                const int* __restrict__ tgt,
                                float* __restrict__ out) {
  int row = blockIdx.x;
  const float* l = logits + (size_t)row * kV;
  float m = -3.4e38f;
  for (int c = threadIdx.x; c < kV; c += blockDim.x) m = fmaxf(m, l[c]);
  m = blockMax(m);
  float s = 0.f;
  for (int c = threadIdx.x; c < kV; c += blockDim.x) s += __expf(l[c] - m);
  s = blockSum(s);
  if (threadIdx.x == 0) out[row] = (m + __logf(s)) - l[tgt[row]];
}

__global__ void mean_kernel(const float* __restrict__ v, float* __restrict__ out) {
  float s = 0.f;
  for (int i = threadIdx.x; i < kM; i += blockDim.x) s += v[i];
  s = blockSum(s);
  if (threadIdx.x == 0) *out = s * (1.f / kM);
}

// =================================================================== launch
extern "C" void kernel_launch(void* const* d_in, const int* in_sizes, int n_in,
                              void* d_out, int out_size, void* d_ws, size_t ws_size,
                              hipStream_t stream) {
  const int*   idx     = (const int*)d_in[0];
  const int*   targets = (const int*)d_in[1];
  const float* tok_emb = (const float*)d_in[2];
  const float* pos_emb = (const float*)d_in[3];
  const float* r_emb   = (const float*)d_in[4];
  const float* r_w     = (const float*)d_in[5];
  const float* r_r     = (const float*)d_in[6];
  const float* qkv_w   = (const float*)d_in[7];
  const float* fc_w    = (const float*)d_in[8];
  const float* ln1_g   = (const float*)d_in[9];
  const float* ln1_b   = (const float*)d_in[10];
  const float* ln2_g   = (const float*)d_in[11];
  const float* ln2_b   = (const float*)d_in[12];
  const float* ffn_w1  = (const float*)d_in[13];
  const float* ffn_b1  = (const float*)d_in[14];
  const float* ffn_w2  = (const float*)d_in[15];
  const float* ffn_b2  = (const float*)d_in[16];
  const float* lnf_g   = (const float*)d_in[17];
  const float* lnf_b   = (const float*)d_in[18];
  const float* head_w  = (const float*)d_in[19];
  const float* head_b  = (const float*)d_in[20];

  float* logits = (float*)d_out;                 // (2048 x 512)
  float* nll_out = logits + (size_t)kM * kV;     // scalar

  char* wsp = (char*)d_ws;
  size_t off = 0;
  auto alloc = [&](size_t bytes) -> void* {
    void* p = wsp + off;
    off += (bytes + 255) & ~(size_t)255;
    return p;
  };
  float* xf   = (float*)alloc((size_t)kM * kEP * 4);    // residual stream
  bf_t*  hbf  = (bf_t*)alloc((size_t)kM * kEP * 2);     // LN output (bf16)
  float* qkvf = (float*)alloc((size_t)kM * kQKVP * 4);
  bf_t*  qwb  = (bf_t*)alloc((size_t)kBH * kT * kDP * 2);
  bf_t*  qrb  = (bf_t*)alloc((size_t)kBH * kT * kDP * 2);
  bf_t*  kbb  = (bf_t*)alloc((size_t)kBH * kT * kDP * 2);
  bf_t*  vTb  = (bf_t*)alloc((size_t)kBH * kDP * kT * 2);
  float* Sb   = (float*)alloc((size_t)kBH * kT * kT * 4);
  bf_t*  Pb   = (bf_t*)alloc((size_t)kBH * kT * kT * 2);
  float* ofb  = (float*)alloc((size_t)kBH * kT * kDP * 4);
  bf_t*  obf  = (bf_t*)alloc((size_t)kM * kEP * 2);
  float* tmpE = (float*)alloc((size_t)kM * kEP * 4);
  float* ff1f = (float*)alloc((size_t)kM * kFFP * 4);
  bf_t*  ff1b = (bf_t*)alloc((size_t)kM * kFFP * 2);
  bf_t*  wbuf = (bf_t*)alloc((size_t)kFFP * kEP * 2);   // reused weight buffer
  bf_t*  rbf  = (bf_t*)alloc((size_t)kRN * kDP * 2);
  float* nllr = (float*)alloc((size_t)kM * 4);

  auto cdiv = [](size_t a, size_t b) { return (unsigned)((a + b - 1) / b); };
  const dim3 blk256(256);

  // embedding + r_emb (bf16, padded)
  embed_kernel<<<cdiv((size_t)kM * kEP, 256), blk256, 0, stream>>>(idx, tok_emb, pos_emb, xf);
  remb_kernel<<<cdiv((size_t)kRN * kDP, 256), blk256, 0, stream>>>(r_emb, rbf);

  for (int l = 0; l < kL; ++l) {
    const float* qw_l = qkv_w + (size_t)l * kE * kQKVN;
    const float* fc_l = fc_w + (size_t)l * kE * kE;
    const float* w1_l = ffn_w1 + (size_t)l * kE * kFF;
    const float* w2_l = ffn_w2 + (size_t)l * kFF * kE;

    // ---- LN1 -> bf16
    ln_bf16_kernel<<<kM, blk256, 0, stream>>>(xf, ln1_g + (size_t)l * kE,
                                              ln1_b + (size_t)l * kE, hbf);
    // ---- qkv = h @ W_qkv
    convT_kernel<<<cdiv((size_t)kQKVP * kEP, 256), blk256, 0, stream>>>(
        qw_l, wbuf, kE, kQKVN, kEP, kQKVP);
    gemm_bf16_128x128<<<dim3(kQKVP / 128, kM / 128, 1), blk256, 0, stream>>>(
        hbf, wbuf, qkvf, kEP, kEP, kEP, kQKVP, 0, 0, 0);
    // ---- split heads, add rel biases, transpose v
    split_qkv_kernel<<<cdiv((size_t)kBH * kT * kDP, 256), blk256, 0, stream>>>(
        qkvf, r_w, r_r, qwb, qrb, kbb, vTb);
    // ---- AC: S[bh] = (q + r_w_bias) k^T   (batched 40x: 512x512x256)
    gemm_bf16_128x128<<<dim3(kT / 128, kT / 128, kBH), blk256, 0, stream>>>(
        qwb, kbb, Sb, kDP, kDP, kDP, kT,
        (long long)kT * kDP, (long long)kT * kDP, (long long)kT * kT);
    // ---- BD: S += (q + r_r_bias) . R[x]
    bd_wmma_kernel<<<dim3(kT / 64, kT), dim3(128), 0, stream>>>(qrb, rbf, Sb);
    // ---- softmax over keys
    softmax_kernel<<<kBH * kT, blk256, 0, stream>>>(Sb, Pb);
    // ---- o[bh] = P v  (batched 40x: 512x256x512)
    gemm_bf16_128x128<<<dim3(kDP / 128, kT / 128, kBH), blk256, 0, stream>>>(
        Pb, vTb, ofb, kT, kT, kT, kDP,
        (long long)kT * kT, (long long)kDP * kT, (long long)kT * kDP);
    merge_o_kernel<<<cdiv((size_t)kM * kEP, 256), blk256, 0, stream>>>(ofb, obf);
    // ---- attn out projection + residual
    convT_kernel<<<cdiv((size_t)kEP * kEP, 256), blk256, 0, stream>>>(
        fc_l, wbuf, kE, kE, kEP, kEP);
    gemm_bf16_128x128<<<dim3(kEP / 128, kM / 128, 1), blk256, 0, stream>>>(
        obf, wbuf, tmpE, kEP, kEP, kEP, kEP, 0, 0, 0);
    add_kernel<<<cdiv((size_t)kM * kEP, 256), blk256, 0, stream>>>(xf, tmpE,
                                                                   (size_t)kM * kEP);
    // ---- LN2 -> bf16
    ln_bf16_kernel<<<kM, blk256, 0, stream>>>(xf, ln2_g + (size_t)l * kE,
                                              ln2_b + (size_t)l * kE, hbf);
    // ---- FFN up + ReLU
    convT_kernel<<<cdiv((size_t)kFFP * kEP, 256), blk256, 0, stream>>>(
        w1_l, wbuf, kE, kFF, kEP, kFFP);
    gemm_bf16_128x128<<<dim3(kFFP / 128, kM / 128, 1), blk256, 0, stream>>>(
        hbf, wbuf, ff1f, kEP, kEP, kEP, kFFP, 0, 0, 0);
    bias_relu_bf16_kernel<<<cdiv((size_t)kM * kFFP, 256), blk256, 0, stream>>>(
        ff1f, ffn_b1 + (size_t)l * kFF, ff1b);
    // ---- FFN down + bias + residual
    convT_kernel<<<cdiv((size_t)kEP * kFFP, 256), blk256, 0, stream>>>(
        w2_l, wbuf, kFF, kE, kFFP, kEP);
    gemm_bf16_128x128<<<dim3(kEP / 128, kM / 128, 1), blk256, 0, stream>>>(
        ff1b, wbuf, tmpE, kFFP, kFFP, kFFP, kEP, 0, 0, 0);
    add_bias_res_kernel<<<cdiv((size_t)kM * kEP, 256), blk256, 0, stream>>>(
        xf, tmpE, ffn_b2 + (size_t)l * kE);
  }

  // ---- final LN, head, loss
  ln_bf16_kernel<<<kM, blk256, 0, stream>>>(xf, lnf_g, lnf_b, hbf);
  convT_kernel<<<cdiv((size_t)kV * kEP, 256), blk256, 0, stream>>>(
      head_w, wbuf, kE, kV, kEP, kV);
  gemm_bf16_128x128<<<dim3(kV / 128, kM / 128, 1), blk256, 0, stream>>>(
      hbf, wbuf, logits, kEP, kEP, kEP, kV, 0, 0, 0);
  head_bias_kernel<<<cdiv((size_t)kM * kV, 256), blk256, 0, stream>>>(logits, head_b);
  nll_rows_kernel<<<kM, blk256, 0, stream>>>(logits, targets, nllr);
  mean_kernel<<<1, blk256, 0, stream>>>(nllr, nll_out);
}
